// DecoderOnlyTransformer_82652350644699
// MI455X (gfx1250) — matrix operand order repaired and checked by reading, not compiled, not run on hardware
//
#include <hip/hip_runtime.h>

// ---------------- types ----------------
typedef __attribute__((ext_vector_type(16))) __bf16        v16bf;
typedef __attribute__((ext_vector_type(8)))  float         v8f;
typedef __attribute__((ext_vector_type(4)))  unsigned int  u32x4;
typedef __attribute__((ext_vector_type(2)))  unsigned int  u32x2;
typedef unsigned short u16;

#define B_   2
#define S_   2048
#define D_   1024
#define H_   16
#define HD_  64
#define FF_  4096
#define V_   32000
#define L_   4

// native f32 -> bf16 (backend emits v_cvt_{pk_}bf16_f32)
__device__ __forceinline__ u16 f2bf(float f) {
    union { __bf16 b; u16 u; } v;
    v.b = (__bf16)f;
    return v.u;
}
__device__ __forceinline__ unsigned pack_bf2(float a, float b) {
    union { __bf16 b[2]; unsigned u; } v;
    v.b[0] = (__bf16)a;
    v.b[1] = (__bf16)b;
    return v.u;
}

union ABFrag { u32x4 q[2]; v16bf h; };

__device__ __forceinline__ v8f wmma_bf16(v16bf a, v16bf b, v8f c) {
    // D = A(16x32 bf16) * B(32x16 bf16) + C(16x16 f32)
    return __builtin_amdgcn_wmma_f32_16x16x32_bf16(false, a, false, b, (short)0, c,
                                                   false, false);
}

// ---------------- fp32 -> bf16 cast (weights), 4 elems/thread ----------------
__global__ __launch_bounds__(256) void cast_bf16_kernel(const float* __restrict__ in,
                                                        u16* __restrict__ out, long n) {
    long i = ((long)blockIdx.x * 256 + threadIdx.x) * 4;
    if (i < n) {  // n % 4 == 0 for all arrays here
        float4 v = *(const float4*)(in + i);
        u32x2 p;
        p[0] = pack_bf2(v.x, v.y);
        p[1] = pack_bf2(v.z, v.w);
        *(u32x2*)(out + i) = p;
    }
}

// ---------------- embedding gather ----------------
__global__ __launch_bounds__(256) void embed_kernel(const int* __restrict__ ids,
                                                    const float* __restrict__ emb,
                                                    float* __restrict__ X) {
    int row = blockIdx.x;                 // b*S + s
    int id  = ids[row];
    int i   = threadIdx.x * 4;
    float4 v = *(const float4*)(emb + (size_t)id * D_ + i);
    *(float4*)(X + (size_t)row * D_ + i) = v;
}

// ---------------- RMSNorm -> bf16 ----------------
__global__ __launch_bounds__(256) void rmsnorm_kernel(const float* __restrict__ X,
                                                      const float* __restrict__ scale,
                                                      u16* __restrict__ Out) {
    int row = blockIdx.x;
    int i   = threadIdx.x * 4;
    float4 v = *(const float4*)(X + (size_t)row * D_ + i);
    float ss = v.x * v.x + v.y * v.y + v.z * v.z + v.w * v.w;
    __shared__ float red[256];
    red[threadIdx.x] = ss;
    __syncthreads();
    for (int s = 128; s > 0; s >>= 1) {
        if (threadIdx.x < s) red[threadIdx.x] += red[threadIdx.x + s];
        __syncthreads();
    }
    float r = rsqrtf(red[0] * (1.0f / D_) + 1e-6f);
    float4 sc = *(const float4*)(scale + i);
    u32x2 p;
    p[0] = pack_bf2(v.x * r * sc.x, v.y * r * sc.y);
    p[1] = pack_bf2(v.z * r * sc.z, v.w * r * sc.w);
    *(u32x2*)(Out + (size_t)row * D_ + i) = p;
}

// ---------------- RoPE + qkv split (q pre-scaled by 1/sqrt(HD)) ----------------
__global__ __launch_bounds__(256) void rope_kernel(const float* __restrict__ QKV, // [B*S, 3*D]
                                                   u16* __restrict__ Qo,          // [B,H,S,HD]
                                                   u16* __restrict__ Ko,          // [B,H,S,HD]
                                                   u16* __restrict__ Vt) {        // [B,H,HD,S]
    int idx = blockIdx.x * 4 + (threadIdx.x >> 6);  // row over B*H*S
    int hd  = threadIdx.x & 63;
    int s   = idx % S_;
    int bh  = idx / S_;
    int h   = bh % H_;
    int b   = bh / H_;
    size_t base = ((size_t)(b * S_ + s)) * (3 * D_) + h * HD_;
    float q = QKV[base + hd];
    float k = QKV[base + D_ + hd];
    float v = QKV[base + 2 * D_ + hd];
    int   prt = (hd < 32) ? hd + 32 : hd - 32;
    float qr  = QKV[base + prt];
    float kr  = QKV[base + D_ + prt];
    float sgn = (hd < 32) ? -1.f : 1.f;
    int   fi  = (hd < 32) ? hd : hd - 32;
    float freq = powf(10000.f, -2.f * (float)fi / (float)HD_);
    float ang  = (float)s * freq;
    float cs = cosf(ang), sn = sinf(ang);
    float qo = q * cs + sgn * qr * sn;
    float ko = k * cs + sgn * kr * sn;
    size_t qk = ((size_t)bh * S_ + s) * HD_ + hd;
    Qo[qk] = f2bf(qo * 0.125f);   // fold 1/sqrt(64)
    Ko[qk] = f2bf(ko);
    Vt[((size_t)bh * HD_ + hd) * S_ + s] = f2bf(v);
}

// ---------------- tiled bf16 GEMM: C = A[M,K] * W[N,K]^T ----------------
// Double-buffered fragments (k unrolled by 64, K % 64 == 0 always here).
// EPI 0: store f32   EPI 1: residual += (f32)   EPI 2: silu -> bf16
template <int EPI>
__global__ __launch_bounds__(256) void gemm_bf16_nt(const u16* __restrict__ A,
                                                    const u16* __restrict__ W,
                                                    void* __restrict__ C,
                                                    int M, int N, int K) {
    int lane = threadIdx.x & 31;
    int wave = threadIdx.x >> 5;
    int l16  = lane & 15;
    int hf   = lane >> 4;
    int wy   = wave >> 2;                 // 0..1  (M dir, 64 rows each)
    int wx   = wave & 3;                  // 0..3  (N dir, 32 cols each)
    int rowBase = blockIdx.y * 128 + wy * 64;
    int colBase = blockIdx.x * 128 + wx * 32;

    const u16* aBase[4];
    const u16* wBase[2];
#pragma unroll
    for (int mt = 0; mt < 4; ++mt)
        aBase[mt] = A + (size_t)(rowBase + mt * 16 + l16) * K + hf * 8;
#pragma unroll
    for (int nt = 0; nt < 2; ++nt)
        wBase[nt] = W + (size_t)(colBase + nt * 16 + l16) * K + hf * 8;

    v8f zero8 = {0.f, 0.f, 0.f, 0.f, 0.f, 0.f, 0.f, 0.f};
    v8f acc[4][2];
#pragma unroll
    for (int mt = 0; mt < 4; ++mt)
#pragma unroll
        for (int nt = 0; nt < 2; ++nt) acc[mt][nt] = zero8;

    ABFrag afr0[4], afr1[4], bfr0[2], bfr1[2];
#pragma unroll
    for (int nt = 0; nt < 2; ++nt) {
        bfr0[nt].q[0] = *(const u32x4*)(wBase[nt]);
        bfr0[nt].q[1] = *(const u32x4*)(wBase[nt] + 16);
    }
#pragma unroll
    for (int mt = 0; mt < 4; ++mt) {
        afr0[mt].q[0] = *(const u32x4*)(aBase[mt]);
        afr0[mt].q[1] = *(const u32x4*)(aBase[mt] + 16);
    }

    for (int k0 = 0; k0 < K; k0 += 64) {
        // stage next half-step (always in range: K % 64 == 0)
#pragma unroll
        for (int nt = 0; nt < 2; ++nt) {
            const u16* p = wBase[nt] + k0 + 32;
            bfr1[nt].q[0] = *(const u32x4*)(p);
            bfr1[nt].q[1] = *(const u32x4*)(p + 16);
        }
#pragma unroll
        for (int mt = 0; mt < 4; ++mt) {
            const u16* p = aBase[mt] + k0 + 32;
            afr1[mt].q[0] = *(const u32x4*)(p);
            afr1[mt].q[1] = *(const u32x4*)(p + 16);
        }
#pragma unroll
        for (int mt = 0; mt < 4; ++mt)
#pragma unroll
            for (int nt = 0; nt < 2; ++nt)
                acc[mt][nt] = wmma_bf16(afr0[mt].h, bfr0[nt].h, acc[mt][nt]);

        if (k0 + 64 < K) {
#pragma unroll
            for (int nt = 0; nt < 2; ++nt) {
                const u16* p = wBase[nt] + k0 + 64;
                bfr0[nt].q[0] = *(const u32x4*)(p);
                bfr0[nt].q[1] = *(const u32x4*)(p + 16);
            }
#pragma unroll
            for (int mt = 0; mt < 4; ++mt) {
                const u16* p = aBase[mt] + k0 + 64;
                afr0[mt].q[0] = *(const u32x4*)(p);
                afr0[mt].q[1] = *(const u32x4*)(p + 16);
            }
        }
#pragma unroll
        for (int mt = 0; mt < 4; ++mt)
#pragma unroll
            for (int nt = 0; nt < 2; ++nt)
                acc[mt][nt] = wmma_bf16(afr1[mt].h, bfr1[nt].h, acc[mt][nt]);
    }

#pragma unroll
    for (int mt = 0; mt < 4; ++mt)
#pragma unroll
        for (int nt = 0; nt < 2; ++nt)
#pragma unroll
            for (int r = 0; r < 8; ++r) {
                int row = rowBase + mt * 16 + r + hf * 8;
                int col = colBase + nt * 16 + l16;
                size_t idx = (size_t)row * N + col;
                float v = acc[mt][nt][r];
                if (EPI == 0) {
                    ((float*)C)[idx] = v;
                } else if (EPI == 1) {
                    float* X = (float*)C;
                    X[idx] += v;
                } else {
                    float sg = 1.f / (1.f + __expf(-v));
                    ((u16*)C)[idx] = f2bf(v * sg);
                }
            }
}

// ---------------- flash attention (causal, online softmax, WMMA) ----------------
// grid: (S/64, B*H), block: 128 (4 waves, 16 q-rows each)
// Dataflow: S-tile (C-layout) -> LDS -> row-layout per lane (8 cols of one row).
// In row-layout, mask/exp/reductions are mostly local VALU; exp'd values are
// already in A-fragment layout for the P*V WMMA (no second LDS pass for P).
__global__ __launch_bounds__(128) void attn_flash_kernel(const u16* __restrict__ Q,   // [BH,S,HD]
                                                         const u16* __restrict__ Kk,  // [BH,S,HD]
                                                         const u16* __restrict__ Vt,  // [BH,HD,S]
                                                         u16* __restrict__ O) {       // [B*S, D]
    __shared__ __align__(16) float lds_s[4][256];  // 16x16 S tile per wave
    __shared__ __align__(16) float lds_c[4][16];   // per-row stat bounce
    int lane = threadIdx.x & 31;
    int w    = threadIdx.x >> 5;
    int l16  = lane & 15;
    int hf   = lane >> 4;
    int bh   = blockIdx.y;
    int qb   = blockIdx.x * 64 + w * 16;          // wave's q base row
    int nsteps = (qb >> 4) + 1;                   // causal: key tiles 0..qb/16
    int qrow_l = qb + l16;                        // row this lane owns in row-layout

    // Q fragments (A layout), K-dim = 64 in two 32-chunks
    ABFrag qf0, qf1;
    {
        const u16* qp = Q + ((size_t)bh * S_ + qb + l16) * HD_;
        qf0.q[0] = *(const u32x4*)(qp + hf * 8);
        qf0.q[1] = *(const u32x4*)(qp + 16 + hf * 8);
        qf1.q[0] = *(const u32x4*)(qp + 32 + hf * 8);
        qf1.q[1] = *(const u32x4*)(qp + 48 + hf * 8);
    }

    float m_row = -__builtin_inff();
    float l_row = 0.f;
    v8f zero8 = {0.f, 0.f, 0.f, 0.f, 0.f, 0.f, 0.f, 0.f};
    v8f of[4];
#pragma unroll
    for (int t = 0; t < 4; ++t) of[t] = zero8;

    u32x4 zq = {0u, 0u, 0u, 0u};
    ABFrag pf; pf.q[0] = zq; pf.q[1] = zq;

    for (int kb = 0; kb < nsteps; ++kb) {
        // ---- S tile = Q * K^T (scale folded into Q) ----
        ABFrag kf0, kf1;
        const u16* kp = Kk + ((size_t)bh * S_ + kb * 16 + l16) * HD_;
        kf0.q[0] = *(const u32x4*)(kp + hf * 8);
        kf0.q[1] = *(const u32x4*)(kp + 16 + hf * 8);
        kf1.q[0] = *(const u32x4*)(kp + 32 + hf * 8);
        kf1.q[1] = *(const u32x4*)(kp + 48 + hf * 8);
        v8f s = zero8;
        s = wmma_bf16(qf0.h, kf0.h, s);
        s = wmma_bf16(qf1.h, kf1.h, s);

        // ---- C-layout -> row-layout via LDS ----
#pragma unroll
        for (int r = 0; r < 8; ++r)
            lds_s[w][(r + hf * 8) * 16 + l16] = s[r];
        asm volatile("s_wait_dscnt 0" ::: "memory");
        float4 sa = *(const float4*)&lds_s[w][l16 * 16 + hf * 8];
        float4 sb = *(const float4*)&lds_s[w][l16 * 16 + hf * 8 + 4];
        float sv[8] = {sa.x, sa.y, sa.z, sa.w, sb.x, sb.y, sb.z, sb.w};

        // ---- causal mask (keys kb*16 + hf*8 + j vs this lane's q row) ----
        int kc = kb * 16 + hf * 8;
#pragma unroll
        for (int j = 0; j < 8; ++j)
            if (kc + j > qrow_l) sv[j] = -__builtin_inff();

        // ---- online softmax: local 8-max + one cross-half shuffle ----
        float mx = sv[0];
#pragma unroll
        for (int j = 1; j < 8; ++j) mx = fmaxf(mx, sv[j]);
        mx = fmaxf(mx, __shfl_xor(mx, 16, 32));
        float mnew = fmaxf(m_row, mx);
        float cor  = __expf(m_row - mnew);
        m_row = mnew;
        float p[8], ls = 0.f;
#pragma unroll
        for (int j = 0; j < 8; ++j) { p[j] = __expf(sv[j] - mnew); ls += p[j]; }
        ls += __shfl_xor(ls, 16, 32);
        l_row = l_row * cor + ls;

        // ---- P already in A-fragment layout: pack bf16 pairs in registers ----
        u32x4 pk;
        pk[0] = pack_bf2(p[0], p[1]);
        pk[1] = pack_bf2(p[2], p[3]);
        pk[2] = pack_bf2(p[4], p[5]);
        pk[3] = pack_bf2(p[6], p[7]);
        pf.q[kb & 1] = pk;

        // ---- redistribute cor to C-layout rows via 16-float LDS bounce ----
        if (hf == 0) lds_c[w][l16] = cor;
        asm volatile("s_wait_dscnt 0" ::: "memory");
        float4 c0 = *(const float4*)&lds_c[w][hf * 8];
        float4 c1 = *(const float4*)&lds_c[w][hf * 8 + 4];
        float corv[8] = {c0.x, c0.y, c0.z, c0.w, c1.x, c1.y, c1.z, c1.w};
#pragma unroll
        for (int t = 0; t < 4; ++t)
#pragma unroll
            for (int r = 0; r < 8; ++r) of[t][r] *= corv[r];

        // ---- P*V flush every 32 keys ----
        bool flush = (kb & 1) || (kb == nsteps - 1);
        if (flush) {
            if (!(kb & 1)) pf.q[1] = zq;          // unpaired last tile: upper half = 0
            int pbase = (kb & ~1) * 16;
            int kaddr = pbase + hf * 8;
            if (kaddr > S_ - 16) kaddr = S_ - 16; // clamp (P is zero there anyway)
#pragma unroll
            for (int t = 0; t < 4; ++t) {
                ABFrag vf;
                const u16* vp = Vt + ((size_t)bh * HD_ + t * 16 + l16) * S_ + kaddr;
                vf.q[0] = *(const u32x4*)(vp);
                vf.q[1] = *(const u32x4*)(vp + 16);
                of[t] = wmma_bf16(pf.h, vf.h, of[t]);
            }
        }
    }

    // ---- epilogue: O / l, write bf16 [B,S,D] ----
    if (hf == 0) lds_c[w][l16] = l_row;
    asm volatile("s_wait_dscnt 0" ::: "memory");
    float4 c0 = *(const float4*)&lds_c[w][hf * 8];
    float4 c1 = *(const float4*)&lds_c[w][hf * 8 + 4];
    float lv[8] = {c0.x, c0.y, c0.z, c0.w, c1.x, c1.y, c1.z, c1.w};
    int b = bh / H_, h = bh % H_;
#pragma unroll
    for (int t = 0; t < 4; ++t)
#pragma unroll
        for (int r = 0; r < 8; ++r) {
            int qrow = qb + r + hf * 8;
            float val = of[t][r] / lv[r];
            O[((size_t)(b * S_ + qrow)) * D_ + h * HD_ + t * 16 + l16] = f2bf(val);
        }
}

// ---------------- host ----------------
extern "C" void kernel_launch(void* const* d_in, const int* in_sizes, int n_in,
                              void* d_out, int out_size, void* d_ws, size_t ws_size,
                              hipStream_t stream) {
    const int*   ids   = (const int*)d_in[0];
    const float* emb   = (const float*)d_in[1];
    const float* qkv_w = (const float*)d_in[2];
    const float* out_w = (const float*)d_in[3];
    const float* fc1_w = (const float*)d_in[4];
    const float* fc2_w = (const float*)d_in[5];
    const float* n1_s  = (const float*)d_in[6];
    const float* n2_s  = (const float*)d_in[7];
    const float* nf_s  = (const float*)d_in[8];
    const float* lm_w  = (const float*)d_in[9];
    float* logits = (float*)d_out;

    char* ws = (char*)d_ws;
    size_t off = 0;
    auto alloc = [&](size_t bytes) -> char* {
        char* p = ws + off;
        off += (bytes + 255) & ~(size_t)255;
        return p;
    };
    const int ROWS = B_ * S_;  // 4096
    u16*   wqkv  = (u16*)alloc((size_t)L_ * 3 * D_ * D_ * 2);
    u16*   wout  = (u16*)alloc((size_t)L_ * D_ * D_ * 2);
    u16*   wfc1  = (u16*)alloc((size_t)L_ * FF_ * D_ * 2);
    u16*   wfc2  = (u16*)alloc((size_t)L_ * D_ * FF_ * 2);
    u16*   wlm   = (u16*)alloc((size_t)V_ * D_ * 2);
    float* x     = (float*)alloc((size_t)ROWS * D_ * 4);
    u16*   hb    = (u16*)alloc((size_t)ROWS * D_ * 2);
    float* qkvb  = (float*)alloc((size_t)ROWS * 3 * D_ * 4);
    u16*   qbuf  = (u16*)alloc((size_t)B_ * H_ * S_ * HD_ * 2);
    u16*   kbuf  = (u16*)alloc((size_t)B_ * H_ * S_ * HD_ * 2);
    u16*   vtbuf = (u16*)alloc((size_t)B_ * H_ * HD_ * S_ * 2);
    u16*   attnb = (u16*)alloc((size_t)ROWS * D_ * 2);
    u16*   ffb   = (u16*)alloc((size_t)ROWS * FF_ * 2);

    // convert weights to bf16 (4 elems per thread)
    long n;
    n = (long)L_ * 3 * D_ * D_;
    cast_bf16_kernel<<<(n / 4 + 255) / 256, 256, 0, stream>>>(qkv_w, wqkv, n);
    n = (long)L_ * D_ * D_;
    cast_bf16_kernel<<<(n / 4 + 255) / 256, 256, 0, stream>>>(out_w, wout, n);
    n = (long)L_ * FF_ * D_;
    cast_bf16_kernel<<<(n / 4 + 255) / 256, 256, 0, stream>>>(fc1_w, wfc1, n);
    n = (long)L_ * D_ * FF_;
    cast_bf16_kernel<<<(n / 4 + 255) / 256, 256, 0, stream>>>(fc2_w, wfc2, n);
    n = (long)V_ * D_;
    cast_bf16_kernel<<<(n / 4 + 255) / 256, 256, 0, stream>>>(lm_w, wlm, n);

    embed_kernel<<<ROWS, 256, 0, stream>>>(ids, emb, x);

    for (int l = 0; l < L_; ++l) {
        rmsnorm_kernel<<<ROWS, 256, 0, stream>>>(x, n1_s + l * D_, hb);
        gemm_bf16_nt<0><<<dim3(3 * D_ / 128, ROWS / 128), 256, 0, stream>>>(
            hb, wqkv + (size_t)l * 3 * D_ * D_, qkvb, ROWS, 3 * D_, D_);
        rope_kernel<<<(B_ * H_ * S_) / 4, 256, 0, stream>>>(qkvb, qbuf, kbuf, vtbuf);
        attn_flash_kernel<<<dim3(S_ / 64, B_ * H_), 128, 0, stream>>>(qbuf, kbuf, vtbuf, attnb);
        gemm_bf16_nt<1><<<dim3(D_ / 128, ROWS / 128), 256, 0, stream>>>(
            attnb, wout + (size_t)l * D_ * D_, x, ROWS, D_, D_);
        rmsnorm_kernel<<<ROWS, 256, 0, stream>>>(x, n2_s + l * D_, hb);
        gemm_bf16_nt<2><<<dim3(FF_ / 128, ROWS / 128), 256, 0, stream>>>(
            hb, wfc1 + (size_t)l * FF_ * D_, ffb, ROWS, FF_, D_);
        gemm_bf16_nt<1><<<dim3(D_ / 128, ROWS / 128), 256, 0, stream>>>(
            ffb, wfc2 + (size_t)l * D_ * FF_, x, ROWS, D_, FF_);
    }

    rmsnorm_kernel<<<ROWS, 256, 0, stream>>>(x, nf_s, hb);
    gemm_bf16_nt<0><<<dim3(V_ / 128, ROWS / 128), 256, 0, stream>>>(
        hb, wlm, logits, ROWS, V_, D_);
}
